// MultiHeadAttention_45286135169093
// MI455X (gfx1250) — matrix operand order repaired and checked by reading, not compiled, not run on hardware
//
#include <hip/hip_runtime.h>
#include <hip/hip_bf16.h>

#define NUM_HEADS 16
#define EMB 1024
#define HEAD 64
#define SEQ 2048
#define BATCH 4
#define HD (NUM_HEADS * HEAD)   // 1024

typedef __bf16 bf16;
typedef __attribute__((ext_vector_type(16))) __bf16 v16bf;
typedef __attribute__((ext_vector_type(8)))  __bf16 v8bf;
typedef __attribute__((ext_vector_type(8)))  float  v8f;
typedef __attribute__((ext_vector_type(4)))  int    v4i;

// ---- gfx1250 async global->LDS path (guarded; falls back to load+store) ----
#if defined(__gfx1250__)
#  if __has_builtin(__builtin_amdgcn_global_load_async_to_lds_b128) && \
      __has_builtin(__builtin_amdgcn_s_wait_asynccnt)
#    define HAVE_ASYNC 1
#  endif
#endif

// copy 16 bytes (8 bf16) global -> LDS
__device__ __forceinline__ void async_copy_b128(const bf16* g, bf16* l) {
#ifdef HAVE_ASYNC
  __builtin_amdgcn_global_load_async_to_lds_b128(
      (__attribute__((address_space(1))) v4i*)g,
      (__attribute__((address_space(3))) v4i*)l, 0, 0);
#else
  *(v8bf*)l = *(const v8bf*)g;
  __builtin_prefetch(g + 64, 0, 1);
#endif
}

__device__ __forceinline__ void wait_async() {
#ifdef HAVE_ASYNC
  __builtin_amdgcn_s_wait_asynccnt(0);
#endif
}

__device__ __forceinline__ v8f wmma_bf16(v16bf a, v16bf b, v8f c) {
  return __builtin_amdgcn_wmma_f32_16x16x32_bf16(false, a, false, b, (short)0, c, false, false);
}

// A fragment (16x32 bf16, M rows x K cols), row-major source with 'stride' elements/row.
// Lane l holds row M = l&15; elements 0..7 = K {0..7 | 8..15}, elements 8..15 = K {16..23 | 24..31}.
__device__ __forceinline__ v16bf load_afrag(const bf16* base, int stride, int lane) {
  const int m   = lane & 15;
  const int off = (lane < 16) ? 0 : 8;
  const bf16* r = base + m * stride;
  v8bf lo = *(const v8bf*)(r + off);
  v8bf hi = *(const v8bf*)(r + 16 + off);
  return __builtin_shufflevector(lo, hi, 0,1,2,3,4,5,6,7,8,9,10,11,12,13,14,15);
}

// B fragment (32x16 bf16, K rows x N cols) sourced from a B^T-layout tile:
// base points at [n][k] tile (16 rows of N, 32 cols of K), 'stride' elements/row.
// Lane l holds column N = l&15; elements 0..7 = K {0..7 | 8..15}, elements 8..15 = K {16..23 | 24..31}.
__device__ __forceinline__ v16bf load_bfrag(const bf16* base, int stride, int lane) {
  const int n   = lane & 15;
  const int off = (lane < 16) ? 0 : 8;
  const bf16* r = base + n * stride;
  v8bf lo = *(const v8bf*)(r + off);
  v8bf hi = *(const v8bf*)(r + 16 + off);
  return __builtin_shufflevector(lo, hi, 0,1,2,3,4,5,6,7,8,9,10,11,12,13,14,15);
}

// ---------------- cast kernels ----------------
__global__ void cast_copy_bf16(const float* __restrict__ in, bf16* __restrict__ out, int n) {
  int i = blockIdx.x * blockDim.x + threadIdx.x;
  if (i < n) out[i] = (bf16)in[i];
}

// in: [H][E][D] f32  ->  out: [H][D][E] bf16
__global__ void cast_transpose_w(const float* __restrict__ in, bf16* __restrict__ out) {
  int i = blockIdx.x * blockDim.x + threadIdx.x;   // total H*E*D = 1M
  int h = i >> 16;                                  // E*D = 65536
  int rem = i & 65535;
  int e = rem >> 6;
  int d = rem & 63;
  out[((h << 6) + d) * EMB + e] = (bf16)in[i];
}

// ---------------- fused QKV projection ----------------
// grid (SEQ/128, H, B), block 256 (8 waves). Each wave: 16 rows, q/k/v 16x64 accum.
// Double-buffered LDS tiles, async copy issue overlapped with WMMA compute.
__global__ void __launch_bounds__(256)
qkv_kernel(const bf16* __restrict__ xb, const bf16* __restrict__ wqT,
           const bf16* __restrict__ wkT, const bf16* __restrict__ wvT,
           bf16* __restrict__ qg, bf16* __restrict__ kg, bf16* __restrict__ vtg) {
  __shared__ __align__(64) bf16 Xs[2][128 * 32];
  __shared__ __align__(64) bf16 Wqs[2][64 * 32];
  __shared__ __align__(64) bf16 Wks[2][64 * 32];
  __shared__ __align__(64) bf16 Wvs[2][64 * 32];

  const int s0 = blockIdx.x * 128;
  const int h  = blockIdx.y;
  const int b  = blockIdx.z;
  const int lane = threadIdx.x & 31;
  const int w    = threadIdx.x >> 5;
  const int w16  = w * 16;

  const bf16* xg = xb + (size_t)b * SEQ * EMB;
  const size_t wbase = (size_t)h * HEAD * EMB;

  auto issue_tiles = [&](int kk, int buf) {
#pragma unroll
    for (int it = 0; it < 2; it++) {
      int c = threadIdx.x + it * 256;
      int row = c >> 2, cc = c & 3;
      async_copy_b128(xg + (size_t)(s0 + row) * EMB + kk + cc * 8,
                      &Xs[buf][row * 32 + cc * 8]);
    }
    {
      int c = threadIdx.x;
      int row = c >> 2, cc = c & 3;
      size_t o = wbase + (size_t)row * EMB + kk + cc * 8;
      int lo = row * 32 + cc * 8;
      async_copy_b128(wqT + o, &Wqs[buf][lo]);
      async_copy_b128(wkT + o, &Wks[buf][lo]);
      async_copy_b128(wvT + o, &Wvs[buf][lo]);
    }
  };

  v8f cq[4], ck[4], cv[4];
#pragma unroll
  for (int n = 0; n < 4; n++) { cq[n] = 0.f; ck[n] = 0.f; cv[n] = 0.f; }

  issue_tiles(0, 0);
  const int nks = EMB / 32;
  for (int ks = 0; ks < nks; ks++) {
    wait_async();
    __syncthreads();                     // current buffer ready for all waves
    if (ks + 1 < nks) issue_tiles((ks + 1) * 32, (ks + 1) & 1);
    const int buf = ks & 1;

    v16bf a = load_afrag(&Xs[buf][w16 * 32], 32, lane);
#pragma unroll
    for (int n = 0; n < 4; n++) {
      v16bf bq = load_bfrag(&Wqs[buf][n * 16 * 32], 32, lane);
      cq[n] = wmma_bf16(a, bq, cq[n]);
      v16bf bk = load_bfrag(&Wks[buf][n * 16 * 32], 32, lane);
      ck[n] = wmma_bf16(a, bk, ck[n]);
      v16bf bv = load_bfrag(&Wvs[buf][n * 16 * 32], 32, lane);
      cv[n] = wmma_bf16(a, bv, cv[n]);
    }
  }

  // epilogue: Q,K -> [b][h][s][d] bf16 ; V -> transposed [b][h][d][s]
  const size_t bhSD = ((size_t)b * NUM_HEADS + h) * SEQ * HEAD;
  const size_t bhDS = ((size_t)b * NUM_HEADS + h) * HEAD * SEQ;
  const int m = lane & 15;
  const int half8 = (lane < 16) ? 0 : 8;
#pragma unroll
  for (int n = 0; n < 4; n++) {
#pragma unroll
    for (int e = 0; e < 8; e++) {
      int s = s0 + w16 + e + half8;
      int d = n * 16 + m;
      qg[bhSD + (size_t)s * HEAD + d]  = (bf16)cq[n][e];
      kg[bhSD + (size_t)s * HEAD + d]  = (bf16)ck[n][e];
      vtg[bhDS + (size_t)d * SEQ + s]  = (bf16)cv[n][e];
    }
  }
}

// ---------------- causal flash attention ----------------
// grid (SEQ/64, H, B), block 128 (4 waves). Each wave: 16 query rows.
// K/V blocks double-buffered; next block prefetched asynchronously during compute.
__global__ void __launch_bounds__(128)
attn_kernel(const bf16* __restrict__ qg, const bf16* __restrict__ kg,
            const bf16* __restrict__ vtg, bf16* __restrict__ cc) {
  __shared__ __align__(64) bf16 Ks[2][64 * 64];
  __shared__ __align__(64) bf16 Vts[2][64 * 64];
  __shared__ __align__(64) bf16 Ps[4 * 16 * 64];

  const int q0  = blockIdx.x * 64;
  const int h   = blockIdx.y;
  const int b   = blockIdx.z;
  const int lane = threadIdx.x & 31;
  const int w    = threadIdx.x >> 5;
  const int w16  = w * 16;
  const int m    = lane & 15;
  const int half8 = (lane < 16) ? 0 : 8;

  const size_t bhSD = ((size_t)b * NUM_HEADS + h) * SEQ * HEAD;
  const size_t bhDS = ((size_t)b * NUM_HEADS + h) * HEAD * SEQ;

  auto issue_kv = [&](int j0, int buf) {
#pragma unroll
    for (int it = 0; it < 4; it++) {
      int c = threadIdx.x + it * 128;
      int row = c >> 3, cc8 = c & 7;
      async_copy_b128(kg + bhSD + (size_t)(j0 * 64 + row) * HEAD + cc8 * 8,
                      &Ks[buf][row * 64 + cc8 * 8]);
      async_copy_b128(vtg + bhDS + (size_t)row * SEQ + j0 * 64 + cc8 * 8,
                      &Vts[buf][row * 64 + cc8 * 8]);
    }
  };

  // load Q fragments for this wave's 16 rows (resident for whole kernel)
  v16bf aq[2];
#pragma unroll
  for (int t = 0; t < 2; t++)
    aq[t] = load_afrag(qg + bhSD + (size_t)(q0 + w16) * HEAD + t * 32, HEAD, lane);

  v8f acc[4];
  float m_r[8], l_r[8];
#pragma unroll
  for (int n = 0; n < 4; n++) acc[n] = 0.f;
#pragma unroll
  for (int e = 0; e < 8; e++) { m_r[e] = -3.0e38f; l_r[e] = 0.f; }

  const int nblk = blockIdx.x + 1;   // causal: only key blocks with k0 <= q0
  issue_kv(0, 0);
  for (int j0 = 0; j0 < nblk; j0++) {
    wait_async();
    __syncthreads();                       // K/V block ready for all waves
    if (j0 + 1 < nblk) issue_kv(j0 + 1, (j0 + 1) & 1);
    const int buf = j0 & 1;

    // scores: Q (16x64) @ K^T (64 x 64kj)
    v8f sc[4];
#pragma unroll
    for (int n = 0; n < 4; n++) sc[n] = 0.f;
#pragma unroll
    for (int t = 0; t < 2; t++) {
#pragma unroll
      for (int n = 0; n < 4; n++) {
        v16bf bk = load_bfrag(&Ks[buf][(n * 16) * 64 + t * 32], 64, lane);
        sc[n] = wmma_bf16(aq[t], bk, sc[n]);
      }
    }

    // scale + causal mask (only diagonal block needs masking)
#pragma unroll
    for (int n = 0; n < 4; n++)
#pragma unroll
      for (int e = 0; e < 8; e++) sc[n][e] = sc[n][e] * 0.125f;
    if (j0 == blockIdx.x) {
#pragma unroll
      for (int n = 0; n < 4; n++)
#pragma unroll
        for (int e = 0; e < 8; e++) {
          int qi = w16 + e + half8;
          int ki = n * 16 + m;
          if (ki > qi) sc[n][e] = -3.0e38f;
        }
    }

    // online softmax: row stats live per (VGPR e, lane-half)
    float mnew[8], alpha[8];
#pragma unroll
    for (int e = 0; e < 8; e++) {
      float v = fmaxf(fmaxf(sc[0][e], sc[1][e]), fmaxf(sc[2][e], sc[3][e]));
#pragma unroll
      for (int msk = 1; msk < 16; msk <<= 1) v = fmaxf(v, __shfl_xor(v, msk, 16));
      mnew[e]  = fmaxf(m_r[e], v);
      alpha[e] = __expf(m_r[e] - mnew[e]);
      m_r[e]   = mnew[e];
    }

    float p[4][8];
#pragma unroll
    for (int n = 0; n < 4; n++)
#pragma unroll
      for (int e = 0; e < 8; e++) {
        p[n][e] = __expf(sc[n][e] - mnew[e]);
        Ps[w * (16 * 64) + (e + half8) * 64 + n * 16 + m] = (bf16)p[n][e];
      }

#pragma unroll
    for (int e = 0; e < 8; e++) {
      float v = (p[0][e] + p[1][e]) + (p[2][e] + p[3][e]);
#pragma unroll
      for (int msk = 1; msk < 16; msk <<= 1) v += __shfl_xor(v, msk, 16);
      l_r[e] = l_r[e] * alpha[e] + v;
    }
#pragma unroll
    for (int n = 0; n < 4; n++)
#pragma unroll
      for (int e = 0; e < 8; e++) acc[n][e] = acc[n][e] * alpha[e];

    // O += P (16x64kj) @ V (64kj x 64d) ; per-wave P region: same-wave LDS RAW is in-order
#pragma unroll
    for (int t = 0; t < 2; t++) {
      v16bf ap = load_afrag(&Ps[w * (16 * 64) + t * 32], 64, lane);
#pragma unroll
      for (int n = 0; n < 4; n++) {
        v16bf bv = load_bfrag(&Vts[buf][(n * 16) * 64 + t * 32], 64, lane);
        acc[n] = wmma_bf16(ap, bv, acc[n]);
      }
    }
  }

  // epilogue: write context directly in concat layout [b][s][h*64+d] (bf16)
#pragma unroll
  for (int e = 0; e < 8; e++) {
    float inv_l = 1.0f / l_r[e];
    int s = q0 + w16 + e + half8;
#pragma unroll
    for (int n = 0; n < 4; n++) {
      cc[((size_t)b * SEQ + s) * HD + h * HEAD + n * 16 + m] = (bf16)(acc[n][e] * inv_l);
    }
  }
}

// ---------------- output projection: out = concat @ Wo^T + bo ----------------
// grid (HD/64, (B*SEQ)/128), block 256 (8 waves). Double-buffered tiles.
__global__ void __launch_bounds__(256)
proj_kernel(const bf16* __restrict__ cc, const bf16* __restrict__ woB,
            const float* __restrict__ bo, float* __restrict__ out) {
  __shared__ __align__(64) bf16 As[2][128 * 32];
  __shared__ __align__(64) bf16 Bs[2][64 * 32];

  const int n0  = blockIdx.x * 64;
  const int m0g = blockIdx.y * 128;
  const int lane = threadIdx.x & 31;
  const int w    = threadIdx.x >> 5;
  const int w16  = w * 16;
  const int m    = lane & 15;
  const int half8 = (lane < 16) ? 0 : 8;

  auto issue_tiles = [&](int kk, int buf) {
#pragma unroll
    for (int it = 0; it < 2; it++) {
      int ci = threadIdx.x + it * 256;
      int row = ci >> 2, cc8 = ci & 3;
      async_copy_b128(cc + (size_t)(m0g + row) * HD + kk + cc8 * 8,
                      &As[buf][row * 32 + cc8 * 8]);
    }
    {
      int ci = threadIdx.x;
      int row = ci >> 2, cc8 = ci & 3;
      // Wo row-major [e_out][hd]: row n of Wo is exactly B^T row n
      async_copy_b128(woB + (size_t)(n0 + row) * HD + kk + cc8 * 8,
                      &Bs[buf][row * 32 + cc8 * 8]);
    }
  };

  v8f c[4];
#pragma unroll
  for (int n = 0; n < 4; n++) c[n] = 0.f;

  issue_tiles(0, 0);
  const int nks = HD / 32;
  for (int ks = 0; ks < nks; ks++) {
    wait_async();
    __syncthreads();
    if (ks + 1 < nks) issue_tiles((ks + 1) * 32, (ks + 1) & 1);
    const int buf = ks & 1;

    v16bf a = load_afrag(&As[buf][w16 * 32], 32, lane);
#pragma unroll
    for (int n = 0; n < 4; n++) {
      v16bf bb = load_bfrag(&Bs[buf][n * 16 * 32], 32, lane);
      c[n] = wmma_bf16(a, bb, c[n]);
    }
  }

#pragma unroll
  for (int n = 0; n < 4; n++) {
    float bias = bo[n0 + n * 16 + m];
#pragma unroll
    for (int e = 0; e < 8; e++) {
      int row = m0g + w16 + e + half8;
      out[(size_t)row * EMB + n0 + n * 16 + m] = c[n][e] + bias;
    }
  }
}

extern "C" void kernel_launch(void* const* d_in, const int* in_sizes, int n_in,
                              void* d_out, int out_size, void* d_ws, size_t ws_size,
                              hipStream_t stream) {
  const float* x  = (const float*)d_in[0];
  const float* Wq = (const float*)d_in[1];
  const float* Wk = (const float*)d_in[2];
  const float* Wv = (const float*)d_in[3];
  const float* Wo = (const float*)d_in[4];
  const float* bo = (const float*)d_in[5];
  float* out = (float*)d_out;

  // workspace layout (bf16 elements)
  bf16* xb  = (bf16*)d_ws;                              // B*S*E      = 8388608
  bf16* wqT = xb  + (size_t)BATCH * SEQ * EMB;          // H*D*E      = 1048576
  bf16* wkT = wqT + (size_t)NUM_HEADS * HEAD * EMB;
  bf16* wvT = wkT + (size_t)NUM_HEADS * HEAD * EMB;
  bf16* woB = wvT + (size_t)NUM_HEADS * HEAD * EMB;     // E*HD       = 1048576
  bf16* qg  = woB + (size_t)EMB * HD;                   // B*H*S*D    = 8388608
  bf16* kg  = qg  + (size_t)BATCH * NUM_HEADS * SEQ * HEAD;
  bf16* vtg = kg  + (size_t)BATCH * NUM_HEADS * SEQ * HEAD;
  bf16* ccb = vtg + (size_t)BATCH * NUM_HEADS * SEQ * HEAD;

  // 1) casts
  {
    int n = BATCH * SEQ * EMB;
    cast_copy_bf16<<<(n + 255) / 256, 256, 0, stream>>>(x, xb, n);
  }
  cast_transpose_w<<<(NUM_HEADS * EMB * HEAD) / 256, 256, 0, stream>>>(Wq, wqT);
  cast_transpose_w<<<(NUM_HEADS * EMB * HEAD) / 256, 256, 0, stream>>>(Wk, wkT);
  cast_transpose_w<<<(NUM_HEADS * EMB * HEAD) / 256, 256, 0, stream>>>(Wv, wvT);
  {
    int n = EMB * HD;
    cast_copy_bf16<<<(n + 255) / 256, 256, 0, stream>>>(Wo, woB, n);
  }

  // 2) fused QKV projection
  qkv_kernel<<<dim3(SEQ / 128, NUM_HEADS, BATCH), 256, 0, stream>>>(
      xb, wqT, wkT, wvT, qg, kg, vtg);

  // 3) causal flash attention
  attn_kernel<<<dim3(SEQ / 64, NUM_HEADS, BATCH), 128, 0, stream>>>(qg, kg, vtg, ccb);

  // 4) output projection + bias
  proj_kernel<<<dim3(HD / 64, (BATCH * SEQ) / 128), 256, 0, stream>>>(ccb, woB, bo, out);
}
